// Head_7447473291284
// MI455X (gfx1250) — matrix operand order repaired and checked by reading, not compiled, not run on hardware
//
#include <hip/hip_runtime.h>
#include <hip/hip_bf16.h>
#include <math.h>

// Problem constants (match reference: B=8, T=2048, C=1024, H=64)
constexpr int BB = 8;
constexpr int TT = 2048;
constexpr int CC = 1024;
constexpr int HH = 64;
constexpr int BT = BB * TT;        // 16384 total rows

typedef __attribute__((ext_vector_type(16))) _Float16 v16h;
typedef __attribute__((ext_vector_type(8)))  _Float16 v8h;
typedef __attribute__((ext_vector_type(8)))  float    v8f;
typedef __attribute__((ext_vector_type(4)))  unsigned int v4u;
typedef __attribute__((ext_vector_type(8)))  int v8i;
typedef __attribute__((ext_vector_type(4)))  int v4i;

#define NEG_INF (-__builtin_inff())

// ---------------------------------------------------------------------------
// WMMA f16 operand loaders (wave32, V_WMMA_F32_16X16X32_F16 layouts)
//
// A (16x32, MxK): lane m = lane&15 holds row m for BOTH lane halves; lane half
// selects the K octet:  e in [0,8)  -> k = half*8 + e
//                       e in [8,16) -> k = half*8 + 16 + (e-8)
// Caller passes p already offset by half*8 into the lane's 32-wide K window.
__device__ __forceinline__ v16h load_a_f16(const _Float16* p) {
  v8h lo = *(const v8h*)(p);
  v8h hi = *(const v8h*)(p + 16);
  v16h r;
#pragma unroll
  for (int e = 0; e < 8; ++e) { r[e] = lo[e]; r[e + 8] = hi[e]; }
  return r;
}

// B (32x16, KxN): lane n = lane&15 holds column n; same K order as A:
// e in [0,8) -> k = half*8 + e, e in [8,16) -> k = half*8 + 16 + (e-8).
// Caller passes p already offset by half*8 into the lane's 32-long K column.
__device__ __forceinline__ v16h load_b_f16(const _Float16* p) {
  v8h lo = *(const v8h*)(p);
  v8h hi = *(const v8h*)(p + 16);
  v16h r;
#pragma unroll
  for (int e = 0; e < 8; ++e) { r[e] = lo[e]; r[e + 8] = hi[e]; }
  return r;
}

// ---------------------------------------------------------------------------
// TDM: 2D tile load Global -> LDS via tensor_load_to_lds (6-arg toolchain
// variant: g0, g1, g2, g3, g4, cpol). D# per CDNA5 ISA
// (group0 128b, group1 256b; groups 2/3 zero -> 2D tensor). All sizes in
// 2-byte elements (data_size code 1).
// ---------------------------------------------------------------------------
__device__ __forceinline__ void tdm_load_2d(unsigned int lds_off,
                                            const void* gaddr,
                                            unsigned int row_elems,   // tensor_dim0
                                            unsigned int nrows,       // tensor_dim1
                                            unsigned int row_stride,  // tensor_dim0_stride
                                            unsigned int tile_w,      // tile_dim0
                                            unsigned int tile_h) {    // tile_dim1
  const unsigned long long ga = (unsigned long long)(uintptr_t)gaddr;
  v4u g0;
  g0[0] = 1u;                                          // count=1, user D#
  g0[1] = lds_off;                                     // lds_addr (bytes)
  g0[2] = (unsigned int)(ga & 0xffffffffu);            // global_addr[31:0]
  g0[3] = (unsigned int)((ga >> 32) & 0x01ffffffu)     // global_addr[56:32]
          | (2u << 30);                                // type=2 ("image")
  v8i g1;
  g1[0] = (int)(1u << 16);                             // data_size=1 (2B), wg_mask=0
  g1[1] = (int)((row_elems & 0xffffu) << 16);          // tensor_dim0[15:0]
  g1[2] = (int)(((row_elems >> 16) & 0xffffu)          // tensor_dim0[31:16]
                | ((nrows & 0xffffu) << 16));          // tensor_dim1[15:0]
  g1[3] = (int)(((nrows >> 16) & 0xffffu)              // tensor_dim1[31:16]
                | ((tile_w & 0xffffu) << 16));         // tile_dim0
  g1[4] = (int)(tile_h & 0xffffu);                     // tile_dim1 (tile_dim2=0)
  g1[5] = (int)row_stride;                             // tensor_dim0_stride[31:0]
  g1[6] = 0;                                           // stride hi, dim1_stride
  g1[7] = 0;
  const v4i z4 = {0, 0, 0, 0};
  const v8i z8 = {0, 0, 0, 0, 0, 0, 0, 0};
  __builtin_amdgcn_tensor_load_to_lds(g0, g1, z4, z4, z8, 0 /*cpol*/);
}

// ---------------------------------------------------------------------------
// Kernel 0: weight prep. fp32 W[C,H] -> f16 Wt[mat][H][C] (transposed), so
// the GEMM B-operand becomes contiguous 16B loads. 0.75 MB total, one-shot.
// ---------------------------------------------------------------------------
__global__ void __launch_bounds__(256)
wprep_kernel(const float* __restrict__ Wq, const float* __restrict__ Wk,
             const float* __restrict__ Wv, _Float16* __restrict__ wt) {
  const int idx = blockIdx.x * 256 + threadIdx.x;     // [0, 3*H*C)
  if (idx >= 3 * HH * CC) return;
  const int mat = idx / (HH * CC);
  const int rem = idx % (HH * CC);
  const int h = rem / CC;
  const int c = rem % CC;
  const float* W = (mat == 0) ? Wq : (mat == 1) ? Wk : Wv;
  wt[idx] = (_Float16)W[(size_t)c * HH + h];
}

// ---------------------------------------------------------------------------
// Kernel 1: fused QKV projection, one pass over x (x read once from HBM).
// One wave: 16-row tile of ALL THREE 64-wide outputs; 12 accumulators,
// 32 K-chunks -> 384 WMMAs per wave.
//   qb, kb : f16 [BT, H]      (row-major, feeds Q A-operand / K^T B-operand)
//   vt     : f16 [B][H][T]    (transposed, feeds PV B-operand contiguously)
// ---------------------------------------------------------------------------
__global__ void __launch_bounds__(128)
qkv_proj_kernel(const float* __restrict__ x, const _Float16* __restrict__ wt,
                _Float16* __restrict__ qb, _Float16* __restrict__ kb,
                _Float16* __restrict__ vt) {
  const int lane = threadIdx.x & 31;
  const int wid  = threadIdx.x >> 5;
  const int lm   = lane & 15;
  const int half = lane >> 4;

  const int rt = blockIdx.x * 4 + wid;        // row tile 0..1023
  const float* xrow = x + (size_t)(rt * 16 + lm) * CC;

  v8f acc[3][4] = {};

  for (int kk = 0; kk < CC; kk += 32) {
    // ---- A operand: x tile, fp32 -> f16 on the fly (4 x 16B loads)
    const float* ap = xrow + kk + half * 8;
    float4 f0 = *(const float4*)(ap);
    float4 f1 = *(const float4*)(ap + 4);
    float4 f2 = *(const float4*)(ap + 16);
    float4 f3 = *(const float4*)(ap + 20);
    v16h a;
    a[0]  = (_Float16)f0.x; a[1]  = (_Float16)f0.y;
    a[2]  = (_Float16)f0.z; a[3]  = (_Float16)f0.w;
    a[4]  = (_Float16)f1.x; a[5]  = (_Float16)f1.y;
    a[6]  = (_Float16)f1.z; a[7]  = (_Float16)f1.w;
    a[8]  = (_Float16)f2.x; a[9]  = (_Float16)f2.y;
    a[10] = (_Float16)f2.z; a[11] = (_Float16)f2.w;
    a[12] = (_Float16)f3.x; a[13] = (_Float16)f3.y;
    a[14] = (_Float16)f3.z; a[15] = (_Float16)f3.w;

    // ---- B operands from pre-transposed f16 weights: contiguous 16B loads
#pragma unroll
    for (int mat = 0; mat < 3; ++mat)
#pragma unroll
      for (int ct = 0; ct < 4; ++ct) {
        const _Float16* wp =
            wt + (size_t)(mat * HH + ct * 16 + lm) * CC + kk + half * 8;
        v16h bm = load_b_f16(wp);
        acc[mat][ct] = __builtin_amdgcn_wmma_f32_16x16x32_f16(
            false, a, false, bm, (short)0, acc[mat][ct], false, false);
      }
  }

  // ---- Q, K stores: C-layout -> f16 [BT, H]
#pragma unroll
  for (int ct = 0; ct < 4; ++ct)
#pragma unroll
    for (int r = 0; r < 8; ++r) {
      const size_t o = (size_t)(rt * 16 + half * 8 + r) * HH + ct * 16 + lm;
      qb[o] = (_Float16)acc[0][ct][r];
      kb[o] = (_Float16)acc[1][ct][r];
    }

  // ---- V store TRANSPOSED: lane's 8 accumulator rows are contiguous in t,
  //      so each column tile is one packed 16B store into Vt[B][H][T].
  const int row0 = rt * 16 + half * 8;        // global t of acc[..][0]
  const int bidx = row0 / TT;                 // 8-run never crosses batch
  const int tl   = row0 % TT;
#pragma unroll
  for (int ct = 0; ct < 4; ++ct) {
    v8h pv;
#pragma unroll
    for (int r = 0; r < 8; ++r) pv[r] = (_Float16)acc[2][ct][r];
    *(v8h*)(vt + (size_t)(bidx * HH + ct * 16 + lm) * TT + tl) = pv;
  }
}

// ---------------------------------------------------------------------------
// Kernel 2: causal flash attention. 4 waves/block = 4 adjacent q-tiles of one
// batch; K/V key-tiles staged into LDS ONCE per block by the Tensor Data
// Mover (wave 0 issues, s_wait_tensorcnt, workgroup barrier publishes).
// Blocks run q-tile group g then 31-g for causal load balance.
// ---------------------------------------------------------------------------
#define LDSP 40   // halfs per P row in LDS: 32 + 8 pad, keeps 16B alignment

__device__ __forceinline__ void attn_group(
    const _Float16* __restrict__ qb, const _Float16* __restrict__ kb,
    const _Float16* __restrict__ vt, float* __restrict__ out,
    int b, int g, int wid, int lm, int half,
    _Float16* Ks, _Float16* Vs, _Float16* pl) {
  const size_t rowOff = (size_t)b * TT;
  const int qbase = (g * 4 + wid) * 16;

  // Q as A-operand, preloaded for both 32-wide K chunks of H=64
  const _Float16* qrow = qb + (rowOff + qbase + lm) * HH;
  const v16h aq0 = load_a_f16(qrow + half * 8);
  const v16h aq1 = load_a_f16(qrow + 32 + half * 8);

  float m_i[8], l_i[8];
  v8f acc[4] = {};
#pragma unroll
  for (int r = 0; r < 8; ++r) { m_i[r] = NEG_INF; l_i[r] = 0.0f; }

  const int nkt  = (qbase >> 5) + 1;   // this wave's causal key-tile count
  const int nktg = 2 * g + 2;          // block-wide key-tile count
  const unsigned int kOff = (unsigned int)(uintptr_t)Ks;
  const unsigned int vOff = (unsigned int)(uintptr_t)Vs;

  for (int j = 0; j < nktg; ++j) {
    const int kstart = j * 32;

    // ---- TDM: stage K tile (32 keys x 64 hd) and V tile (64 hd x 32 t)
    if (wid == 0) {
      tdm_load_2d(kOff, kb + (rowOff + kstart) * HH,
                  HH, TT, HH, /*tile_w=*/HH, /*tile_h=*/32);
      tdm_load_2d(vOff, vt + (size_t)b * HH * TT + kstart,
                  TT, HH, TT, /*tile_w=*/32, /*tile_h=*/HH);
      __builtin_amdgcn_s_wait_tensorcnt(0);
    }
    __syncthreads();                       // publish LDS tiles to all waves

    if (j < nkt) {
      // ---- S = (Q K^T) * scale, causal-masked, in C-layout registers
      v8f s[2];
#pragma unroll
      for (int sub = 0; sub < 2; ++sub) {
        const int key = kstart + sub * 16 + lm;
        const _Float16* kr = Ks + (sub * 16 + lm) * HH;
        v16h bk0 = load_b_f16(kr + half * 8);
        v16h bk1 = load_b_f16(kr + 32 + half * 8);
        v8f t = {};
        t = __builtin_amdgcn_wmma_f32_16x16x32_f16(
            false, aq0, false, bk0, (short)0, t, false, false);
        t = __builtin_amdgcn_wmma_f32_16x16x32_f16(
            false, aq1, false, bk1, (short)0, t, false, false);
#pragma unroll
        for (int r = 0; r < 8; ++r) {
          float v = t[r] * 0.125f;               // H^-0.5 = 1/8
          if (key > qbase + half * 8 + r) v = NEG_INF;   // causal mask
          t[r] = v;
        }
        s[sub] = t;
      }

      // ---- row max over 32 keys: pair max + 16-lane xor-shuffle tree
      float mnew[8], corr[8];
#pragma unroll
      for (int r = 0; r < 8; ++r) {
        float mx = fmaxf(s[0][r], s[1][r]);
        mx = fmaxf(mx, __shfl_xor(mx, 1, 32));
        mx = fmaxf(mx, __shfl_xor(mx, 2, 32));
        mx = fmaxf(mx, __shfl_xor(mx, 4, 32));
        mx = fmaxf(mx, __shfl_xor(mx, 8, 32));
        mnew[r] = fmaxf(m_i[r], mx);
      }

      // ---- P = exp(S - m); spill P to LDS in f16 for layout transpose
#pragma unroll
      for (int sub = 0; sub < 2; ++sub)
#pragma unroll
        for (int r = 0; r < 8; ++r) {
          float p = __expf(s[sub][r] - mnew[r]); // exp(-inf)=0 handles mask
          s[sub][r] = p;
          pl[(half * 8 + r) * LDSP + sub * 16 + lm] = (_Float16)p;
        }

      // ---- row sum + online rescale of l and O accumulators
#pragma unroll
      for (int r = 0; r < 8; ++r) {
        float sm = s[0][r] + s[1][r];
        sm += __shfl_xor(sm, 1, 32);
        sm += __shfl_xor(sm, 2, 32);
        sm += __shfl_xor(sm, 4, 32);
        sm += __shfl_xor(sm, 8, 32);
        corr[r] = __expf(m_i[r] - mnew[r]);
        l_i[r]  = l_i[r] * corr[r] + sm;
        m_i[r]  = mnew[r];
      }
#pragma unroll
      for (int ct = 0; ct < 4; ++ct)
#pragma unroll
        for (int r = 0; r < 8; ++r) acc[ct][r] *= corr[r];

      // ---- P (C-layout) -> A-layout via LDS; order with CDNA5 DS counter
      asm volatile("s_wait_dscnt 0" ::: "memory");
      const v16h ap = load_a_f16(pl + lm * LDSP + half * 8);

      // ---- O += P @ V from the LDS-staged V tile (contiguous 16B ds reads)
#pragma unroll
      for (int ct = 0; ct < 4; ++ct) {
        v16h bv = load_b_f16(Vs + (ct * 16 + lm) * 32 + half * 8);
        acc[ct] = __builtin_amdgcn_wmma_f32_16x16x32_f16(
            false, ap, false, bv, (short)0, acc[ct], false, false);
      }
    }
    __syncthreads();                 // all reads done before next TDM store
  }

  // ---- epilogue: out = O / l, fp32 [B, T, H]
#pragma unroll
  for (int ct = 0; ct < 4; ++ct)
#pragma unroll
    for (int r = 0; r < 8; ++r)
      out[(rowOff + qbase + half * 8 + r) * HH + ct * 16 + lm] =
          acc[ct][r] / l_i[r];
}

__global__ void __launch_bounds__(128)
flash_attn_kernel(const _Float16* __restrict__ qb,
                  const _Float16* __restrict__ kb,
                  const _Float16* __restrict__ vt,
                  float* __restrict__ out) {
  __shared__ _Float16 Ks[32 * HH];            // 4 KB K tile  [key][hd]
  __shared__ _Float16 Vs[HH * 32];            // 4 KB V tile  [hd][t]
  __shared__ _Float16 ptile[4][16 * LDSP];    // per-wave P transpose

  const int lane = threadIdx.x & 31;
  const int wid  = threadIdx.x >> 5;
  const int lm   = lane & 15;
  const int half = lane >> 4;

  const int b  = blockIdx.x >> 4;             // batch
  const int gp = blockIdx.x & 15;             // group pair index 0..15
  _Float16* pl = &ptile[wid][0];

  // Causal load balance: groups g and 31-g have ~complementary work.
  attn_group(qb, kb, vt, out, b, gp,      wid, lm, half, Ks, Vs, pl);
  attn_group(qb, kb, vt, out, b, 31 - gp, wid, lm, half, Ks, Vs, pl);
}

// ---------------------------------------------------------------------------
// Launch: inputs are {x, Wk, Wq, Wv} (setup_inputs dict order!)
// ---------------------------------------------------------------------------
extern "C" void kernel_launch(void* const* d_in, const int* in_sizes, int n_in,
                              void* d_out, int out_size, void* d_ws, size_t ws_size,
                              hipStream_t stream) {
  (void)in_sizes; (void)n_in; (void)out_size; (void)ws_size;

  const float* x  = (const float*)d_in[0];
  const float* Wk = (const float*)d_in[1];
  const float* Wq = (const float*)d_in[2];
  const float* Wv = (const float*)d_in[3];
  float* out = (float*)d_out;

  // Workspace: qb, kb f16 [BT,H] (2 MB each), vt f16 [B][H][T] (2 MB),
  // wt f16 [3][H][C] (384 KB). All L2-resident (192 MB L2).
  _Float16* qb = (_Float16*)d_ws;
  _Float16* kb = qb + (size_t)BT * HH;
  _Float16* vt = kb + (size_t)BT * HH;
  _Float16* wt = vt + (size_t)BT * HH;

  // Kernel 0: weight transpose+convert (one-shot)
  wprep_kernel<<<dim3((3 * HH * CC + 255) / 256), 256, 0, stream>>>(
      Wq, Wk, Wv, wt);

  // Kernel 1: 1024 row tiles, 4 waves per block, single pass over x
  qkv_proj_kernel<<<dim3(BT / 16 / 4), 128, 0, stream>>>(x, wt, qb, kb, vt);

  // Kernel 2: 8 batches x 16 balanced group pairs = 128 blocks of 4 waves
  flash_attn_kernel<<<dim3(BB * 16), 128, 0, stream>>>(qb, kb, vt, out);
}
